// mamba_SS_model_80255758893611
// MI455X (gfx1250) — compile-verified, matrix operand order checked by reading, not run on hardware
//
#include <hip/hip_runtime.h>

// ---------------- model dims ----------------
#define B_    128
#define CB_   200
#define HW_   32
#define HID_  32
#define P_    4
#define D_    128
#define DI_   256
#define DS_   16
#define DR_   8
#define KC_   4     // conv kernel
#define DEPTH_ 4
#define L_SPA 65
#define L_SPE 101

typedef __attribute__((ext_vector_type(16))) __bf16 v16bf;
typedef __attribute__((ext_vector_type(8)))  float  v8f;

__device__ __forceinline__ float act_apply(float v, int act) {
  if (act == 1) return v > 0.f ? v : 0.f;                       // relu
  if (act == 2) return v > 20.f ? v : log1pf(__expf(v));        // softplus
  if (act == 3) return 1.f / (1.f + __expf(-v));                // sigmoid
  return v;
}

// ---------------------------------------------------------------------------
// WMMA GEMM:  C[M,N] = act( A[M,K] * W[N,K]^T + bias[N] ),  C row-major.
// A address (handles NHWC / patch views, all divisors are powers of two):
//   m -> (m1=m>>aMi1Sh, rem=m&mask1, m2=rem>>aMi2Sh, m3=rem&mask2)
//   k -> (k>>aKiSh)*aKs1 + (k&kmask)*aKs2
//   addr = m1*aS1 + m2*aS2 + m3*aS3 + koff
// Requirements: M % 64 == 0 (true at every call site). K,N arbitrary: clamp
// addresses in-bounds, select-to-zero after the load (all loads unconditional).
// 64x64 block tile, BK=32, 8 waves, double-buffered LDS software pipeline:
// global loads for tile t+1 are issued before the WMMAs on tile t (LOADcnt and
// DScnt are independent), one barrier per K-step.
// f32 -> bf16 conversion into LDS; math = v_wmma_f32_16x16x32_bf16 (f32 accum).
// ---------------------------------------------------------------------------
__global__ void __launch_bounds__(256)
gemm_wmma_k(const float* __restrict__ A, const float* __restrict__ W,
            const float* __restrict__ bias, float* __restrict__ C,
            int M, int N, int K,
            int aMi1Sh, int aMi2Sh, int aS1, int aS2, int aS3,
            int aKiSh, int aKs1, int aKs2,
            int act)
{
  __shared__ __bf16 As[2][64][34];
  __shared__ __bf16 Bs[2][64][34];

  const int tid   = threadIdx.x;
  const int lane  = tid & 31;
  const int wave  = tid >> 5;
  const int lm    = lane & 15;
  const int half  = lane >> 4;
  const int mrow0 = (wave & 3) * 16;   // sub-tile row base (0..48)
  const int ncol0 = (wave >> 2) * 32;  // sub-tile col base (0 or 32)
  const int m_blk = blockIdx.y * 64;
  const int n_blk = blockIdx.x * 64;

  const int ldk  = tid & 31;   // k lane for cooperative loads (coalesced)
  const int ldr0 = tid >> 5;   // starting row
  const int aMi1Mask = (1 << aMi1Sh) - 1;
  const int aMi2Mask = (1 << aMi2Sh) - 1;
  const int aKiMask  = (1 << aKiSh) - 1;

  // Hoist per-row base addresses out of the K loop (shifts only, no division).
  int aRow[8]; int wRow[8]; unsigned wOk = 0;
#pragma unroll
  for (int i = 0; i < 8; ++i) {
    int r  = ldr0 + i * 8;
    int gm = m_blk + r;                       // always < M (M % 64 == 0)
    int m1 = gm >> aMi1Sh; int rem = gm & aMi1Mask;
    int m2 = rem >> aMi2Sh; int m3 = rem & aMi2Mask;
    aRow[i] = m1 * aS1 + m2 * aS2 + m3 * aS3;
    int gn = n_blk + r;
    if (gn < N) wOk |= (1u << i);
    wRow[i] = (gn < N ? gn : N - 1) * K;      // clamped, in-bounds address
  }

  float va[8], vb[8];
  auto load_tile = [&](int k0) {
    int kg = k0 + ldk;
    int kc = kg < K ? kg : K - 1;             // clamped, in-bounds address
    int koff = (kc >> aKiSh) * aKs1 + (kc & aKiMask) * aKs2;
#pragma unroll
    for (int i = 0; i < 8; ++i) va[i] = A[aRow[i] + koff];
#pragma unroll
    for (int i = 0; i < 8; ++i) vb[i] = W[wRow[i] + kc];
  };
  auto store_tile = [&](int p, int k0) {
    bool kOk = (k0 + ldk) < K;
#pragma unroll
    for (int i = 0; i < 8; ++i) {
      int r = ldr0 + i * 8;
      As[p][r][ldk] = (__bf16)(kOk ? va[i] : 0.f);
      Bs[p][r][ldk] = (__bf16)((kOk && (wOk & (1u << i))) ? vb[i] : 0.f);
    }
  };

  v8f acc0 = {}; v8f acc1 = {};

  // pipeline prologue
  load_tile(0);
  store_tile(0, 0);
  __syncthreads();

  int p = 0;
  for (int k0 = 0; k0 < K; k0 += 32) {
    bool more = (k0 + 32) < K;
    if (more) load_tile(k0 + 32);             // overlaps with WMMA below

    // Fragments per ISA layout: lane holds row (lane%16); VGPR v pair:
    // kbase = (v<4?0:16) + (lane/16)*8 + (v%4)*2
    v16bf af, bfa, bfb;
#pragma unroll
    for (int v = 0; v < 8; ++v) {
      int kb = ((v & 4) << 2) + (half << 3) + ((v & 3) << 1);
      af[2*v]    = As[p][mrow0 + lm][kb];
      af[2*v+1]  = As[p][mrow0 + lm][kb + 1];
      bfa[2*v]   = Bs[p][ncol0 + lm][kb];
      bfa[2*v+1] = Bs[p][ncol0 + lm][kb + 1];
      bfb[2*v]   = Bs[p][ncol0 + 16 + lm][kb];
      bfb[2*v+1] = Bs[p][ncol0 + 16 + lm][kb + 1];
    }
    acc0 = __builtin_amdgcn_wmma_f32_16x16x32_bf16(false, af, false, bfa, (short)0, acc0, false, false);
    acc1 = __builtin_amdgcn_wmma_f32_16x16x32_bf16(false, af, false, bfb, (short)0, acc1, false, false);

    if (more) {
      store_tile(p ^ 1, k0 + 32);             // waits LOADcnt here, not earlier
      __syncthreads();                        // single barrier per K-step
      p ^= 1;
    }
  }

  (void)M;
  // Epilogue: C row-major. tile row = i + 8*(lane>>4), col = lane&15
#pragma unroll
  for (int i = 0; i < 8; ++i) {
    int gm = m_blk + mrow0 + i + 8 * half;
    int gn0 = n_blk + ncol0 + lm;
    if (gn0 < N) {
      float v = acc0[i] + (bias ? bias[gn0] : 0.f);
      C[(size_t)gm * N + gn0] = act_apply(v, act);
    }
    int gn1 = gn0 + 16;
    if (gn1 < N) {
      float v = acc1[i] + (bias ? bias[gn1] : 0.f);
      C[(size_t)gm * N + gn1] = act_apply(v, act);
    }
  }
}

// ---------------- data movement / reduction kernels ----------------

// x [B][CB][1024] (NCHW) -> xt [B][1024][CB] (NHWC), 32x32 LDS tiles.
__global__ void transpose_x_k(const float* __restrict__ x, float* __restrict__ xt) {
  __shared__ float t[32][33];
  int b  = blockIdx.z;
  int c0 = blockIdx.y * 32;   // channel tile
  int p0 = blockIdx.x * 32;   // pixel tile
  int tx = threadIdx.x, ty = threadIdx.y;   // (32, 8)
#pragma unroll
  for (int j = 0; j < 32; j += 8) {
    int c = c0 + ty + j, p = p0 + tx;
    t[ty + j][tx] = (c < CB_) ? x[((size_t)b * CB_ + c) * 1024 + p] : 0.f;
  }
  __syncthreads();
#pragma unroll
  for (int j = 0; j < 32; j += 8) {
    int p = p0 + ty + j, c = c0 + tx;
    if (c < CB_) xt[((size_t)b * 1024 + p) * CB_ + c] = t[tx][ty + j];
  }
}

// Single-pass BN statistics over [M, HID] buffer: 128 partial blocks, then a
// deterministic finalize (no float atomics -> bit-stable across replays).
__global__ void bn_partial_k(const float* __restrict__ h, float* __restrict__ part, int M) {
  __shared__ float sh[256], shq[256];
  int tid = threadIdx.x; int c = tid & 31; int r0 = tid >> 5;
  float s = 0.f, q = 0.f;
  for (int m = blockIdx.x * 8 + r0; m < M; m += gridDim.x * 8) {
    float v = h[(size_t)m * HID_ + c]; s += v; q += v * v;
  }
  sh[tid] = s; shq[tid] = q; __syncthreads();
  for (int k = 128; k >= 32; k >>= 1) {
    if (tid < k) { sh[tid] += sh[tid + k]; shq[tid] += shq[tid + k]; }
    __syncthreads();
  }
  if (tid < 32) {
    part[blockIdx.x * 64 + tid] = sh[tid];
    part[blockIdx.x * 64 + 32 + tid] = shq[tid];
  }
}

__global__ void bn_final_k(const float* __restrict__ part, const float* __restrict__ g,
                           const float* __restrict__ b, float* __restrict__ stat,
                           int M, int nblk) {
  int c = threadIdx.x;
  if (c >= HID_) return;
  float s = 0.f, q = 0.f;
  for (int i = 0; i < nblk; ++i) { s += part[i * 64 + c]; q += part[i * 64 + 32 + c]; }
  float mean = s / (float)M;
  float var  = q / (float)M - mean * mean;
  float sc = g[c] * rsqrtf(var + 1e-5f);
  stat[c] = sc;
  stat[HID_ + c] = b[c] - mean * sc;
}

__global__ void bn_apply_k(float* h, const float* __restrict__ stat, int n, int relu) {
  int i = blockIdx.x * 256 + threadIdx.x;
  if (i >= n) return;
  int c = i & (HID_ - 1);
  float v = h[i] * stat[c] + stat[HID_ + c];
  if (relu) v = fmaxf(v, 0.f);
  h[i] = v;
}

// spa_pw [d][c][p][q] -> [d][(p,q,c)] so patch-embed K dim matches NHWC strides
__global__ void permute_spa_pw_k(const float* __restrict__ w, float* __restrict__ o) {
  int i = blockIdx.x * 256 + threadIdx.x;
  if (i >= D_ * 512) return;
  int d = i >> 9, k = i & 511;
  int p = k >> 7, rem = k & 127, q = rem >> 5, c = rem & 31;
  o[i] = w[(((size_t)d * HID_ + c) * P_ + p) * P_ + q];
}

// Gather 5x5 center crop into contiguous A [B*100, 50]  (k = c*2+s)
__global__ void pack_spe_k(const float* __restrict__ x, float* __restrict__ a) {
  int i = blockIdx.x * 256 + threadIdx.x;
  const int n = B_ * 100 * 50;
  if (i >= n) return;
  int k = i % 50, m = i / 50;
  int l = m % 100, b = m / 100;
  int s = k & 1, c = k >> 1;
  int row = 14 + c / 5, col = 14 + c % 5;
  a[i] = x[(((size_t)b * CB_ + (2 * l + s)) * HW_ + row) * HW_ + col];
}

// Scatter contiguous embed output src[B,(L-1),D] into token buffer x[B,L,D]
// with positional embedding; last token = cls + pos.
__global__ void addpos_cls_k(const float* __restrict__ src, float* __restrict__ x,
                             const float* __restrict__ pos, const float* __restrict__ cls,
                             int L) {
  int i = blockIdx.x * 256 + threadIdx.x;
  int n = B_ * L * D_;
  if (i >= n) return;
  int d = i & (D_ - 1);
  int t = (i >> 7) % L;
  int b = i / (L * D_);
  if (t == L - 1) x[i] = cls[d] + pos[t * D_ + d];
  else            x[i] = src[((size_t)(b * (L - 1) + t)) * D_ + d] + pos[t * D_ + d];
}

__global__ void layernorm_k(const float* __restrict__ x, const float* __restrict__ g,
                            const float* __restrict__ b, float* __restrict__ o) {
  __shared__ float red[D_];
  int t = blockIdx.x, d = threadIdx.x;
  float v = x[(size_t)t * D_ + d];
  red[d] = v; __syncthreads();
  for (int s = D_ / 2; s > 0; s >>= 1) { if (d < s) red[d] += red[d + s]; __syncthreads(); }
  float mean = red[0] / (float)D_;
  __syncthreads();
  float c = v - mean;
  red[d] = c * c; __syncthreads();
  for (int s = D_ / 2; s > 0; s >>= 1) { if (d < s) red[d] += red[d + s]; __syncthreads(); }
  float var = red[0] / (float)D_;
  o[(size_t)t * D_ + d] = c * rsqrtf(var + 1e-6f) * g[d] + b[d];
}

// x2[l] = x[L-2-l] for l<L-1 ; x2[L-1] = x[L-1]
__global__ void reverse_k(const float* __restrict__ in, float* __restrict__ outp, int L) {
  int i = blockIdx.x * 256 + threadIdx.x;
  int n = B_ * L * D_;
  if (i >= n) return;
  int d = i & (D_ - 1);
  int l = (i >> 7) % L;
  int b = i / (L * D_);
  int sl = (l == L - 1) ? l : (L - 2 - l);
  outp[i] = in[((size_t)(b * L + sl)) * D_ + d];
}

// causal depthwise conv (K=4) over the xc half of xz, + SiLU
__global__ void conv_silu_k(const float* __restrict__ xz, const float* __restrict__ cw,
                            const float* __restrict__ cb, float* __restrict__ xc, int L) {
  int i = blockIdx.x * 256 + threadIdx.x;
  int n = B_ * L * DI_;
  if (i >= n) return;
  int d = i & (DI_ - 1);
  int l = (i >> 8) % L;
  int b = i / (L * DI_);
  float acc = cb[d];
#pragma unroll
  for (int j = 0; j < KC_; ++j) {
    int ll = l - (KC_ - 1) + j;
    if (ll >= 0) acc += cw[d * KC_ + j] * xz[((size_t)(b * L + ll)) * (2 * DI_) + d];
  }
  float s = 1.f / (1.f + __expf(-acc));
  xc[i] = acc * s;
}

// selective scan: block per batch element, thread per channel d.
// Fuses: dA=exp(-dt*exp(alog)), h recurrence, y = h.C + xc*D, gate y*silu(z).
__global__ void __launch_bounds__(256)
scan_k(const float* __restrict__ dbl, const float* __restrict__ dt,
       const float* __restrict__ xc,  const float* __restrict__ xz,
       const float* __restrict__ alog, const float* __restrict__ dp,
       float* __restrict__ yg, int L) {
  int b = blockIdx.x, d = threadIdx.x;
  float eA[DS_];
#pragma unroll
  for (int s = 0; s < DS_; ++s) eA[s] = __expf(alog[d * DS_ + s]);
  float dpar = dp[d];
  float h[DS_];
#pragma unroll
  for (int s = 0; s < DS_; ++s) h[s] = 0.f;
  __shared__ float sBC[2 * DS_];
  for (int l = 0; l < L; ++l) {
    size_t t = (size_t)(b * L + l);
    __syncthreads();
    if (d < 2 * DS_) sBC[d] = dbl[t * (DR_ + 2 * DS_) + DR_ + d];
    __syncthreads();
    float dtv = dt[t * DI_ + d];
    float xcv = xc[t * DI_ + d];
    float zv  = xz[t * (2 * DI_) + DI_ + d];
    float dtx = dtv * xcv;
    float y = 0.f;
#pragma unroll
    for (int s = 0; s < DS_; ++s) {
      float a = __expf(-dtv * eA[s]);
      h[s] = a * h[s] + dtx * sBC[s];
      y += h[s] * sBC[DS_ + s];
    }
    y += xcv * dpar;
    float sg = 1.f / (1.f + __expf(-zv));
    yg[t * DI_ + d] = y * (zv * sg);
  }
}

__global__ void combine_k(float* io, const float* __restrict__ a,
                          const float* __restrict__ b, int n) {
  int i = blockIdx.x * 256 + threadIdx.x;
  if (i < n) io[i] += 0.5f * (a[i] + b[i]);
}

__global__ void cross_c_k(const float* __restrict__ xspa, const float* __restrict__ xspe,
                          float* __restrict__ c) {
  int i = blockIdx.x * 256 + threadIdx.x;
  if (i >= B_ * D_) return;
  int b = i >> 7, d = i & (D_ - 1);
  float s = 0.f;
  for (int l = 0; l < L_SPE; ++l) s += xspe[((size_t)(b * L_SPE + l)) * D_ + d];
  s /= (float)L_SPE;
  float mid = xspa[((size_t)(b * L_SPA + (L_SPA - 1) / 2)) * D_ + d];
  c[i] = 0.5f * (mid + s);
}

__global__ void scale_k(float* x, const float* __restrict__ sig, int L) {
  int i = blockIdx.x * 256 + threadIdx.x;
  int n = B_ * L * D_;
  if (i >= n) return;
  int b = i / (L * D_);
  int d = i & (D_ - 1);
  x[i] *= sig[b * D_ + d];
}

__global__ void head_in_k(const float* __restrict__ xspa, const float* __restrict__ xspe,
                          float* __restrict__ h) {
  int i = blockIdx.x * 256 + threadIdx.x;
  if (i >= B_ * D_) return;
  int b = i >> 7, d = i & (D_ - 1);
  h[i] = 0.5f * (xspa[((size_t)(b * L_SPA + L_SPA - 1)) * D_ + d] +
                 xspe[((size_t)(b * L_SPE + L_SPE - 1)) * D_ + d]);
}

// ---------------------------------------------------------------------------
extern "C" void kernel_launch(void* const* d_in, const int* in_sizes, int n_in,
                              void* d_out, int out_size, void* d_ws, size_t ws_size,
                              hipStream_t stream) {
  (void)in_sizes; (void)n_in; (void)out_size; (void)ws_size;
  const float* x       = (const float*)d_in[0];
  const float* spa_pos = (const float*)d_in[1];
  const float* spe_pos = (const float*)d_in[2];
  const float* dr_w1   = (const float*)d_in[3];
  const float* dr_b1   = (const float*)d_in[4];
  const float* bn1_g   = (const float*)d_in[5];
  const float* bn1_b   = (const float*)d_in[6];
  const float* dr_w2   = (const float*)d_in[7];
  const float* dr_b2   = (const float*)d_in[8];
  const float* bn2_g   = (const float*)d_in[9];
  const float* bn2_b   = (const float*)d_in[10];
  const float* spa_pw  = (const float*)d_in[11];
  const float* spa_pb  = (const float*)d_in[12];
  const float* spe_pw  = (const float*)d_in[13];
  const float* spe_pb  = (const float*)d_in[14];
  const float* spa_cls = (const float*)d_in[15];
  const float* spe_cls = (const float*)d_in[16];
  const float* ln_g    = (const float*)d_in[17];
  const float* ln_b    = (const float*)d_in[18];
  const float* in_w    = (const float*)d_in[19];
  const float* conv_w  = (const float*)d_in[20];
  const float* conv_b  = (const float*)d_in[21];
  const float* xp_w    = (const float*)d_in[22];
  const float* dtp_w   = (const float*)d_in[23];
  const float* dtp_b   = (const float*)d_in[24];
  const float* a_log   = (const float*)d_in[25];
  const float* d_p     = (const float*)d_in[26];
  const float* out_w   = (const float*)d_in[27];
  const float* l1_w    = (const float*)d_in[28];
  const float* head_w  = (const float*)d_in[29];
  const float* head_b  = (const float*)d_in[30];
  float* out = (float*)d_out;

  // workspace bump allocator (floats)
  float* wsp = (float*)d_ws;
  auto alloc = [&](size_t n) { float* p = wsp; wsp += n; return p; };
  const int Mconv = B_ * HW_ * HW_;                    // 131072
  float* xt   = alloc((size_t)B_ * 1024 * CB_);        // NHWC input
  float* h1   = alloc((size_t)Mconv * HID_);
  float* h2   = alloc((size_t)Mconv * HID_);
  float* part = alloc(128 * 64);
  float* stat = alloc(64);
  float* pwr  = alloc((size_t)D_ * 512);
  float* aspe = alloc((size_t)B_ * 100 * 50);
  float* emb  = alloc((size_t)B_ * 100 * D_);          // contiguous embed temp
  float* xspa = alloc((size_t)B_ * L_SPA * D_);
  float* xspe = alloc((size_t)B_ * L_SPE * D_);
  float* xn   = alloc((size_t)B_ * L_SPE * D_);
  float* x2b  = alloc((size_t)B_ * L_SPE * D_);
  float* x1b  = alloc((size_t)B_ * L_SPE * D_);
  float* x3b  = alloc((size_t)B_ * L_SPE * D_);
  float* xz   = alloc((size_t)B_ * L_SPE * 2 * DI_);
  float* xcb  = alloc((size_t)B_ * L_SPE * DI_);
  float* dbl  = alloc((size_t)B_ * L_SPE * (DR_ + 2 * DS_));
  float* dtb  = alloc((size_t)B_ * L_SPE * DI_);
  float* ygb  = alloc((size_t)B_ * L_SPE * DI_);
  float* cbuf = alloc((size_t)B_ * D_);
  float* sigb = alloc((size_t)B_ * D_);
  float* hin  = alloc((size_t)B_ * D_);

  auto gemm = [&](const float* A, const float* W, const float* bias, float* C,
                  int M, int N, int K,
                  int aMi1Sh, int aMi2Sh, int aS1, int aS2, int aS3,
                  int aKiSh, int aKs1, int aKs2, int act) {
    dim3 g((N + 63) / 64, (M + 63) / 64);
    gemm_wmma_k<<<g, 256, 0, stream>>>(A, W, bias, C, M, N, K,
                                       aMi1Sh, aMi2Sh, aS1, aS2, aS3,
                                       aKiSh, aKs1, aKs2, act);
  };
  auto gemm_rm = [&](const float* A, const float* W, const float* bias, float* C,
                     int M, int N, int K, int act) {
    gemm(A, W, bias, C, M, N, K, 0, 0, K, 0, 0, 0, 1, 0, act);
  };
  auto n_blocks = [](int n) { return (n + 255) / 256; };
  auto bnorm = [&](float* h, const float* g, const float* b, int relu) {
    bn_partial_k<<<128, 256, 0, stream>>>(h, part, Mconv);
    bn_final_k<<<1, 64, 0, stream>>>(part, g, b, stat, Mconv, 128);
    bn_apply_k<<<n_blocks(Mconv * HID_), 256, 0, stream>>>(h, stat, Mconv * HID_, relu);
  };

  // SSM: xz=in_proj -> causal conv+silu -> xp_w -> dt(softplus) -> scan(gated) -> out_proj
  auto ssm = [&](const float* xin, float* xout, int L, int li, int st) {
    int pi = li * 2 + st;
    const float* p_inw  = in_w  + (size_t)pi * 2 * DI_ * D_;
    const float* p_cw   = conv_w + (size_t)pi * DI_ * KC_;
    const float* p_cb   = conv_b + (size_t)pi * DI_;
    const float* p_xpw  = xp_w  + (size_t)pi * (DR_ + 2 * DS_) * DI_;
    const float* p_dtw  = dtp_w + (size_t)pi * DI_ * DR_;
    const float* p_dtb  = dtp_b + (size_t)pi * DI_;
    const float* p_alog = a_log + (size_t)pi * DI_ * DS_;
    const float* p_dp   = d_p   + (size_t)pi * DI_;
    const float* p_ow   = out_w + (size_t)pi * D_ * DI_;
    int M = B_ * L;
    gemm_rm(xin, p_inw, nullptr, xz, M, 2 * DI_, D_, 0);
    conv_silu_k<<<n_blocks(M * DI_), 256, 0, stream>>>(xz, p_cw, p_cb, xcb, L);
    gemm_rm(xcb, p_xpw, nullptr, dbl, M, DR_ + 2 * DS_, DI_, 0);
    // A = dbl[:, 0:8] (row stride 40), softplus epilogue
    gemm(dbl, p_dtw, p_dtb, dtb, M, DI_, DR_,
         0, 0, DR_ + 2 * DS_, 0, 0, 0, 1, 0, 2);
    scan_k<<<B_, DI_, 0, stream>>>(dbl, dtb, xcb, xz, p_alog, p_dp, ygb, L);
    gemm_rm(ygb, p_ow, nullptr, xout, M, D_, DI_, 0);
  };

  // bidirectional block
  auto blockfn = [&](float* xio, int L, int li, int st) {
    const float* lg = ln_g + (size_t)(li * 2 + st) * D_;
    const float* lb = ln_b + (size_t)(li * 2 + st) * D_;
    int n = B_ * L * D_;
    layernorm_k<<<B_ * L, D_, 0, stream>>>(xio, lg, lb, xn);
    ssm(xn, x1b, L, li, st);
    reverse_k<<<n_blocks(n), 256, 0, stream>>>(xn, x2b, L);
    ssm(x2b, x3b, L, li, st);
    reverse_k<<<n_blocks(n), 256, 0, stream>>>(x3b, xn, L);   // xn = reversed x3
    combine_k<<<n_blocks(n), 256, 0, stream>>>(xio, x1b, xn, n);
  };

  // ---- stage 0: transpose x to NHWC so conv GEMMs are K-contiguous ----
  transpose_x_k<<<dim3(32, (CB_ + 31) / 32, B_), dim3(32, 8), 0, stream>>>(x, xt);

  // ---- stage 1: 1x1 conv + BN + relu ----
  gemm_rm(xt, dr_w1, dr_b1, h1, Mconv, HID_, CB_, 0);
  bnorm(h1, bn1_g, bn1_b, 1);
  // ---- stage 2: 1x1 conv + BN ----
  gemm_rm(h1, dr_w2, dr_b2, h2, Mconv, HID_, HID_, 0);
  bnorm(h2, bn2_g, bn2_b, 0);

  // ---- patch embed (spatial tokens) ----
  permute_spa_pw_k<<<n_blocks(D_ * 512), 256, 0, stream>>>(spa_pw, pwr);
  // A[m=(b,i,j)][k=(p,q,c)] over NHWC h2 -> contiguous emb [B*64, D]
  gemm(h2, pwr, spa_pb, emb, B_ * 64, D_, 512,
       6, 3, 32768, 4096, 128,   7, 1024, 1, 0);
  addpos_cls_k<<<n_blocks(B_ * L_SPA * D_), 256, 0, stream>>>(emb, xspa, spa_pos, spa_cls, L_SPA);

  // ---- spectral embed ----
  pack_spe_k<<<n_blocks(B_ * 100 * 50), 256, 0, stream>>>(x, aspe);
  gemm_rm(aspe, spe_pw, spe_pb, emb, B_ * 100, D_, 50, 0);
  addpos_cls_k<<<n_blocks(B_ * L_SPE * D_), 256, 0, stream>>>(emb, xspe, spe_pos, spe_cls, L_SPE);

  // ---- depth layers ----
  for (int li = 0; li < DEPTH_; ++li) {
    blockfn(xspa, L_SPA, li, 0);
    blockfn(xspe, L_SPE, li, 1);
    cross_c_k<<<n_blocks(B_ * D_), 256, 0, stream>>>(xspa, xspe, cbuf);
    gemm_rm(cbuf, l1_w + (size_t)li * D_ * D_, nullptr, sigb, B_, D_, D_, 3);
    scale_k<<<n_blocks(B_ * L_SPA * D_), 256, 0, stream>>>(xspa, sigb, L_SPA);
    scale_k<<<n_blocks(B_ * L_SPE * D_), 256, 0, stream>>>(xspe, sigb, L_SPE);
  }

  // ---- head ----
  head_in_k<<<n_blocks(B_ * D_), 256, 0, stream>>>(xspa, xspe, hin);
  gemm_rm(hin, head_w, head_b, out, B_, 10, D_, 0);
}
